// OURLSTM_4587025072793
// MI455X (gfx1250) — compile-verified, matrix operand order checked by reading, not run on hardware
//
#include <hip/hip_runtime.h>

typedef __attribute__((ext_vector_type(16))) _Float16 v16h;
typedef __attribute__((ext_vector_type(8)))  float    v8f;

#define F 96
#define KCH 4
#define KTOT 384          // K * F
#define NOUT 288          // 3 live gates (i, t, o) * H
#define NTILES 18         // NOUT / 16
#define KBLOCKS 12        // KTOT / 32

// ---------------------------------------------------------------------------
// small elementwise / graph kernels
// ---------------------------------------------------------------------------
__global__ void zero_f32(float* p, int n) {
    int i = blockIdx.x * blockDim.x + threadIdx.x;
    if (i < n) p[i] = 0.0f;
}

__global__ void deg_kernel(const int* src, const float* w, float* deg, int E) {
    int e = blockIdx.x * blockDim.x + threadIdx.x;
    if (e < E) atomicAdd(&deg[src[e]], w[e]);
}

__global__ void dinv_kernel(const float* deg, float* dinv, int n) {
    int i = blockIdx.x * blockDim.x + threadIdx.x;
    if (i < n) {
        float d = deg[i];
        dinv[i] = (d > 0.0f) ? rsqrtf(d) : 0.0f;
    }
}

__global__ void nw_kernel(const int* src, const int* dst, const float* w,
                          const float* dinv, float* nw, int E) {
    int e = blockIdx.x * blockDim.x + threadIdx.x;
    if (e < E) nw[e] = -w[e] * dinv[src[e]] * dinv[dst[e]];
}

// lap(): out[dst] += nw[e] * in[src]; one thread per (edge, 4-feature group)
__global__ void spmm_kernel(const int* src, const int* dst, const float* nw,
                            const float* vin, float* vout, int total /* E*24 */) {
    int t = blockIdx.x * blockDim.x + threadIdx.x;
    if (t >= total) return;
    int e = t / 24;
    int q = t - e * 24;
    float w = nw[e];
    const float4 v = *(const float4*)(vin + (size_t)src[e] * F + q * 4);
    float* o = vout + (size_t)dst[e] * F + q * 4;
    atomicAdd(o + 0, w * v.x);
    atomicAdd(o + 1, w * v.y);
    atomicAdd(o + 2, w * v.z);
    atomicAdd(o + 3, w * v.w);
}

// Chebyshev recurrence: t = 2*t - s
__global__ void cheb_axpy(float* t, const float* s, int n) {
    int i = blockIdx.x * blockDim.x + threadIdx.x;
    if (i < n) t[i] = 2.0f * t[i] - s[i];
}

// ---------------------------------------------------------------------------
// WMMA fragment packing (ISA 7.12.2: 16-bit A 16x32 per-lane K mapping)
// ---------------------------------------------------------------------------
__device__ __forceinline__ int kmap(int j, int half) {
    int p = j >> 1;
    return 2 * (p & 3) + ((p >> 2) << 4) + (half << 3) + (j & 1);
}

// A fragments: layout [mtile][kb][lane][16] halves; row = mtile*16 + (lane&15)
__global__ void pack_a(const float* x, const float* t1, const float* t2,
                       const float* t3, _Float16* afrag, int nrows, int total) {
    int gid = blockIdx.x * blockDim.x + threadIdx.x;  // (mt*12 + kb)*32 + lane
    if (gid >= total) return;
    int lane = gid & 31;
    int kb   = (gid >> 5) % KBLOCKS;
    int mt   = gid / (KBLOCKS * 32);
    int half = lane >> 4;
    int row  = mt * 16 + (lane & 15);
    v16h av;
    for (int j = 0; j < 16; ++j) {
        int K = kb * 32 + kmap(j, half);
        int poly = K / F, f = K - poly * F;
        float v = 0.0f;
        if (row < nrows) {
            const float* s = (poly == 0) ? x : (poly == 1) ? t1
                           : (poly == 2) ? t2 : t3;
            v = s[(size_t)row * F + f];
        }
        av[j] = (_Float16)v;
    }
    *(v16h*)(afrag + (size_t)gid * 16) = av;
}

// B fragments from Wx [4,K,F,H]: layout [ntile][kb][lane][16]; col = nt*16+(lane&15)
// live gate mapping: out col block 0 -> gate 0 (i), 1 -> gate 2 (c~), 2 -> gate 3 (o)
__global__ void pack_b(const float* Wx, _Float16* bfrag, int total) {
    int gid = blockIdx.x * blockDim.x + threadIdx.x;  // (nt*12 + kb)*32 + lane
    if (gid >= total) return;
    int lane = gid & 31;
    int kb   = (gid >> 5) % KBLOCKS;
    int nt   = gid / (KBLOCKS * 32);
    int half = lane >> 4;
    int n    = nt * 16 + (lane & 15);
    int gsel = n / F;
    int gate = (gsel == 0) ? 0 : (gsel == 1) ? 2 : 3;
    int h = n - gsel * F;
    v16h bv;
    for (int j = 0; j < 16; ++j) {
        int K = kb * 32 + kmap(j, half);
        int k = K / F, f = K - k * F;
        bv[j] = (_Float16)Wx[(((size_t)gate * KCH + k) * F + f) * F + h];
    }
    *(v16h*)(bfrag + (size_t)gid * 16) = bv;
}

// combined bias: bsum[n] = bx[g,h] + bh[g,h] + b_gate[g,h]
__global__ void bsum_kernel(const float* bx, const float* bh, const float* bg,
                            float* bsum) {
    int n = blockIdx.x * blockDim.x + threadIdx.x;
    if (n >= NOUT) return;
    int gsel = n / F;
    int gate = (gsel == 0) ? 0 : (gsel == 1) ? 2 : 3;
    int h = n - gsel * F;
    bsum[n] = bx[gate * F + h] + bh[gate * F + h] + bg[gate * F + h];
}

// ---------------------------------------------------------------------------
// dense GEMM: pre[N, 288] = T[N, 384] @ W'[384, 288]  via v_wmma_f32_16x16x32_f16
// one wave per 16x16 output tile, 12 WMMAs along K
// ---------------------------------------------------------------------------
__global__ void gemm_wmma(const _Float16* __restrict__ afrag,
                          const _Float16* __restrict__ bfrag,
                          float* __restrict__ pre, int nrows) {
    int lane = threadIdx.x;                       // 0..31
    int nt   = blockIdx.y * blockDim.y + threadIdx.y;
    int mt   = blockIdx.x;
    if (nt >= NTILES) return;                     // wave-uniform, never splits EXEC
    const _Float16* ap = afrag + (size_t)mt * KBLOCKS * 512;
    const _Float16* bp = bfrag + (size_t)nt * KBLOCKS * 512;
    v8f acc = {};
    for (int kb = 0; kb < KBLOCKS; ++kb) {
        v16h a = *(const v16h*)(ap + kb * 512 + lane * 16);
        v16h b = *(const v16h*)(bp + kb * 512 + lane * 16);
        acc = __builtin_amdgcn_wmma_f32_16x16x32_f16(
            false, a, false, b, (short)0, acc, false, false);
    }
    int half = lane >> 4;
    int col  = nt * 16 + (lane & 15);
    for (int r = 0; r < 8; ++r) {
        int row = mt * 16 + r + half * 8;
        if (row < nrows) pre[(size_t)row * NOUT + col] = acc[r];
    }
}

// ---------------------------------------------------------------------------
// gate epilogue: one wave per row; lane handles h = lane, lane+32, lane+64
// ---------------------------------------------------------------------------
__global__ void gate_kernel(const float* __restrict__ pre,
                            const float* __restrict__ bsum,
                            const float* __restrict__ wc_o,
                            const float* __restrict__ lin_w,
                            const float* __restrict__ lin_b,
                            float* __restrict__ out, int nrows) {
    int gtid = blockIdx.x * blockDim.x + threadIdx.x;
    int row  = gtid >> 5;
    int lane = gtid & 31;
    if (row >= nrows) return;  // wave-uniform
    const float* p = pre + (size_t)row * NOUT;
    float acc = 0.0f;
    for (int j = 0; j < 3; ++j) {
        int h = lane + 32 * j;
        float ai = p[h]           + bsum[h];
        float at = p[F + h]       + bsum[F + h];
        float ao = p[2 * F + h]   + bsum[2 * F + h];
        float iv = 1.0f / (1.0f + expf(-ai));
        float tv = tanhf(at);
        float c  = iv * tv;
        float ov = 1.0f / (1.0f + expf(-(ao + wc_o[h] * c)));
        float hh = ov * tanhf(c);
        acc += fmaxf(hh, 0.0f) * lin_w[h];
    }
    for (int off = 16; off > 0; off >>= 1) acc += __shfl_xor(acc, off, 32);
    if (lane == 0) out[row] = acc + lin_b[0];
}

// ---------------------------------------------------------------------------
extern "C" void kernel_launch(void* const* d_in, const int* in_sizes, int n_in,
                              void* d_out, int out_size, void* d_ws, size_t ws_size,
                              hipStream_t stream) {
    const float* x   = (const float*)d_in[0];
    const int*   ei  = (const int*)d_in[1];
    const float* ew  = (const float*)d_in[2];
    const float* Wx  = (const float*)d_in[3];
    const float* bx  = (const float*)d_in[4];
    // d_in[5] = Wh: unused (h0 == 0 -> cheb(h) == bh)
    const float* bh  = (const float*)d_in[6];
    const float* wc  = (const float*)d_in[7];
    const float* bg  = (const float*)d_in[8];
    const float* lw  = (const float*)d_in[9];
    const float* lb  = (const float*)d_in[10];

    const int N = in_sizes[0] / F;
    const int E = in_sizes[1] / 2;
    const int* src = ei;
    const int* dst = ei + E;
    const int NF = N * F;
    const int mtiles = (N + 15) / 16;

    // workspace carve-up (256B aligned)
    char* ws = (char*)d_ws;
    size_t off = 0;
    auto carve = [&](size_t bytes) -> char* {
        char* p = ws + off;
        off = (off + bytes + 255) & ~(size_t)255;
        return p;
    };
    float*    deg   = (float*)carve((size_t)N * 4);
    float*    dinv  = (float*)carve((size_t)N * 4);
    float*    nw    = (float*)carve((size_t)E * 4);
    float*    t1    = (float*)carve((size_t)NF * 4);
    float*    t2    = (float*)carve((size_t)NF * 4);
    float*    t3    = (float*)carve((size_t)NF * 4);
    _Float16* afrag = (_Float16*)carve((size_t)mtiles * KBLOCKS * 512 * 2);
    _Float16* bfrag = (_Float16*)carve((size_t)NTILES * KBLOCKS * 512 * 2);
    float*    bsum  = (float*)carve(NOUT * 4);
    float*    pre   = (float*)carve((size_t)N * NOUT * 4);
    float*    out   = (float*)d_out;

    const int B = 256;
    auto blks = [](int n, int b) { return (n + b - 1) / b; };

    // --- graph normalization ---
    zero_f32<<<blks(N, B), B, 0, stream>>>(deg, N);
    deg_kernel<<<blks(E, B), B, 0, stream>>>(src, ew, deg, E);
    dinv_kernel<<<blks(N, B), B, 0, stream>>>(deg, dinv, N);
    nw_kernel<<<blks(E, B), B, 0, stream>>>(src, dst, ew, dinv, nw, E);

    // --- Chebyshev bases: t1 = L x ; t2 = 2 L t1 - x ; t3 = 2 L t2 - t1 ---
    const int spmm_total = E * 24;
    zero_f32<<<blks(NF, B), B, 0, stream>>>(t1, NF);
    spmm_kernel<<<blks(spmm_total, B), B, 0, stream>>>(src, dst, nw, x, t1, spmm_total);
    zero_f32<<<blks(NF, B), B, 0, stream>>>(t2, NF);
    spmm_kernel<<<blks(spmm_total, B), B, 0, stream>>>(src, dst, nw, t1, t2, spmm_total);
    cheb_axpy<<<blks(NF, B), B, 0, stream>>>(t2, x, NF);
    zero_f32<<<blks(NF, B), B, 0, stream>>>(t3, NF);
    spmm_kernel<<<blks(spmm_total, B), B, 0, stream>>>(src, dst, nw, t2, t3, spmm_total);
    cheb_axpy<<<blks(NF, B), B, 0, stream>>>(t3, t1, NF);

    // --- pack WMMA fragments ---
    const int a_total = mtiles * KBLOCKS * 32;
    const int b_total = NTILES * KBLOCKS * 32;
    pack_a<<<blks(a_total, B), B, 0, stream>>>(x, t1, t2, t3, afrag, N, a_total);
    pack_b<<<blks(b_total, B), B, 0, stream>>>(Wx, bfrag, b_total);
    bsum_kernel<<<blks(NOUT, B), B, 0, stream>>>(bx, bh, bg, bsum);

    // --- dense GEMM on the WMMA pipe: pre = T @ W' ---
    dim3 gblock(32, 6);
    dim3 ggrid(mtiles, (NTILES + 5) / 6);
    gemm_wmma<<<ggrid, gblock, 0, stream>>>(afrag, bfrag, pre, N);

    // --- gates + readout ---
    const int gate_threads = N * 32;
    gate_kernel<<<blks(gate_threads, B), B, 0, stream>>>(
        pre, bsum, wc + 2 * F, lw, lb, out, N);
}
